// Atention_79396765433864
// MI455X (gfx1250) — matrix operand order repaired and checked by reading, not compiled, not run on hardware
//
#include <hip/hip_runtime.h>

#define NUM_HEADS 12
#define DIM_HEADS 64
#define DIM    768
#define DH     768
#define BATCH  8
#define SEQ    1024
#define QKV_OUT 2304

typedef __attribute__((ext_vector_type(16))) _Float16 v16h;
typedef __attribute__((ext_vector_type(8)))  float    v8f;
typedef __attribute__((ext_vector_type(4)))  float    v4f;
typedef __attribute__((ext_vector_type(4)))  _Float16 v4h;

static __device__ __forceinline__ v8f wmma_f16(v16h a, v16h b, v8f c) {
  // D = A(16x32 f16) * B(32x16 f16) + C(16x16 f32)
  return __builtin_amdgcn_wmma_f32_16x16x32_f16(false, a, false, b, (short)0, c,
                                                false, false);
}

// ---- fragment loaders for WMMA operands (wave32); A and B share one contiguous k order ----
// A 16x32 f16: lane m=lane&15 holds row m; lane-half kh=lane>>4 selects K = kh*16 + j
// -> one contiguous 32B load per lane.
static __device__ __forceinline__ v16h load_a_f16(const _Float16* p, int ld, int lane) {
  const int m = lane & 15, kh = lane >> 4;
  const _Float16* q = p + (size_t)m * ld + kh * 16;
  v16h a;
#pragma unroll
  for (int j = 0; j < 8; ++j) a[j] = q[j];
#pragma unroll
  for (int j = 0; j < 8; ++j) a[j + 8] = q[8 + j];
  return a;
}
// B 32x16 f16: lane n=lane&15 holds output column n; K = kh*16 + j.
// Source row-major [col][K] -> one contiguous 32B load per lane.
static __device__ __forceinline__ v16h load_b_f16(const _Float16* p, int ld, int lane) {
  const int n = lane & 15, kh = lane >> 4;
  const _Float16* q = p + (size_t)n * ld + kh * 16;
  v16h b;
#pragma unroll
  for (int j = 0; j < 16; ++j) b[j] = q[j];
  return b;
}
// B 32x16 f16 from transposed V: Vt[d][key], K runs over keys (contiguous).
static __device__ __forceinline__ v16h load_b_vt(const _Float16* p, int lane) {
  const int n = lane & 15, kh = lane >> 4;
  const _Float16* q = p + (size_t)n * SEQ + kh * 16;
  v16h b;
#pragma unroll
  for (int j = 0; j < 16; ++j) b[j] = q[j];
  return b;
}

// ---------------- kernel 0: one-shot fp32 -> fp16 conversion ----------------
__global__ void __launch_bounds__(256) cvt_f32_f16(const float* __restrict__ s,
                                                   _Float16* __restrict__ d) {
  const size_t i = (size_t)blockIdx.x * 256 + threadIdx.x;   // one float4 each
  v4f x = ((const v4f*)s)[i];
  v4h y;
  y[0] = (_Float16)x[0]; y[1] = (_Float16)x[1];
  y[2] = (_Float16)x[2]; y[3] = (_Float16)x[3];
  ((v4h*)d)[i] = y;
}

// -------- kernel 1: QKV projection (all-f16 GEMM, 32x64 per wave) --------
// writes q,k as [B,H,N,64] and V transposed as Vt[B,H,64,N]
__global__ void __launch_bounds__(128) qkv_proj_kernel(
    const _Float16* __restrict__ img, const _Float16* __restrict__ Wqkv,
    _Float16* __restrict__ q16, _Float16* __restrict__ k16,
    _Float16* __restrict__ vt16) {
  const int lane = threadIdx.x & 31;
  const int wave = threadIdx.x >> 5;
  const int task = blockIdx.x * 4 + wave;          // 256 * 36 tasks
  const int NG = QKV_OUT / 64;                     // 36 col groups of 64
  const int mt = task / NG;
  const int n0 = (task % NG) * 64;
  const int row0 = mt * 32;

  v8f acc[2][4] = {};
  for (int k0 = 0; k0 < DIM; k0 += 32) {
    const v16h a0 = load_a_f16(img + (size_t)row0 * DIM + k0, DIM, lane);
    const v16h a1 = load_a_f16(img + (size_t)(row0 + 16) * DIM + k0, DIM, lane);
#pragma unroll
    for (int nt = 0; nt < 4; ++nt) {
      const v16h b = load_b_f16(Wqkv + (size_t)(n0 + nt * 16) * DIM + k0, DIM, lane);
      acc[0][nt] = wmma_f16(a0, b, acc[0][nt]);
      acc[1][nt] = wmma_f16(a1, b, acc[1][nt]);
    }
  }
  // C layout: lane l, reg r -> row r + 8*(l>>4), col l&15
  const int rbase = (lane >> 4) << 3;
  const int cn = lane & 15;
#pragma unroll
  for (int nt = 0; nt < 4; ++nt) {
    const int e0 = n0 + nt * 16;
    const int h = e0 / 192;
    const int part = (e0 % 192) / 64;
    const int d0 = e0 % 64;
#pragma unroll
    for (int rs = 0; rs < 2; ++rs) {
#pragma unroll
      for (int r = 0; r < 8; ++r) {
        const int row = row0 + rs * 16 + rbase + r;   // flat b*SEQ + n
        const int bb = row >> 10, nn = row & 1023;
        const _Float16 val = (_Float16)acc[rs][nt][r];
        const size_t head = (size_t)bb * NUM_HEADS + h;
        if (part == 0)
          q16[(head * SEQ + nn) * DIM_HEADS + d0 + cn] = val;
        else if (part == 1)
          k16[(head * SEQ + nn) * DIM_HEADS + d0 + cn] = val;
        else
          vt16[(head * DIM_HEADS + d0 + cn) * SEQ + nn] = val;   // transposed
      }
    }
  }
}

// ---------------- kernel 2: flash attention -> f16 [B*N, 768] ----------------
__global__ void __launch_bounds__(128) attn_kernel(
    const _Float16* __restrict__ q16, const _Float16* __restrict__ k16,
    const _Float16* __restrict__ vt16, _Float16* __restrict__ ao) {
  __shared__ _Float16 lds_p[4][16 * 32];          // per-wave P tile (1 KB each)
  const int lane = threadIdx.x & 31;
  const int wave = threadIdx.x >> 5;
  _Float16* pbuf = lds_p[wave];
  const int task = blockIdx.x * 4 + wave;          // 8*12*64 tasks
  const int QT = SEQ / 16;
  const int qt = task % QT;
  const int h  = (task / QT) % NUM_HEADS;
  const int bb = task / (QT * NUM_HEADS);
  const size_t hoff = ((size_t)bb * NUM_HEADS + h) * SEQ * DIM_HEADS;
  const _Float16* Q  = q16 + hoff;
  const _Float16* K  = k16 + hoff;
  const _Float16* Vt = vt16 + hoff;                // [64][SEQ]
  const int q0 = qt * 16;

  const v16h aq0 = load_a_f16(Q + (size_t)q0 * DIM_HEADS + 0,  DIM_HEADS, lane);
  const v16h aq1 = load_a_f16(Q + (size_t)q0 * DIM_HEADS + 32, DIM_HEADS, lane);

  v8f o[4] = {};
  float mrow[8], lrow[8];
#pragma unroll
  for (int r = 0; r < 8; ++r) { mrow[r] = -3.0e38f; lrow[r] = 0.0f; }

  for (int key0 = 0; key0 < SEQ; key0 += 32) {
    // S tiles: cols key0..+15 and key0+16..+31, reduced over d=64 (two K-steps)
    v8f s0 = {}, s1 = {};
    {
      v16h bk = load_b_f16(K + (size_t)key0 * DIM_HEADS + 0, DIM_HEADS, lane);
      s0 = wmma_f16(aq0, bk, s0);
      bk = load_b_f16(K + (size_t)key0 * DIM_HEADS + 32, DIM_HEADS, lane);
      s0 = wmma_f16(aq1, bk, s0);
      bk = load_b_f16(K + (size_t)(key0 + 16) * DIM_HEADS + 0, DIM_HEADS, lane);
      s1 = wmma_f16(aq0, bk, s1);
      bk = load_b_f16(K + (size_t)(key0 + 16) * DIM_HEADS + 32, DIM_HEADS, lane);
      s1 = wmma_f16(aq1, bk, s1);
    }
    // online softmax: row r+8*(lane>>4) lives entirely in one 16-lane half
#pragma unroll
    for (int r = 0; r < 8; ++r) {
      const float x0 = s0[r] * 8.0f;               // reference multiplies by sqrt(64)
      const float x1 = s1[r] * 8.0f;
      float tmax = fmaxf(x0, x1);
#pragma unroll
      for (int off = 8; off >= 1; off >>= 1)
        tmax = fmaxf(tmax, __shfl_xor(tmax, off, 32));
      const float mnew = fmaxf(mrow[r], tmax);
      const float corr = __expf(mrow[r] - mnew);
      const float p0 = __expf(x0 - mnew);
      const float p1 = __expf(x1 - mnew);
      float tsum = p0 + p1;
#pragma unroll
      for (int off = 8; off >= 1; off >>= 1)
        tsum += __shfl_xor(tsum, off, 32);
      lrow[r] = lrow[r] * corr + tsum;
      mrow[r] = mnew;
#pragma unroll
      for (int dt = 0; dt < 4; ++dt) o[dt][r] *= corr;  // rescale O (same row map)
      const int row = r + ((lane >> 4) << 3);
      const int col = lane & 15;
      pbuf[row * 32 + col] = (_Float16)p0;
      pbuf[row * 32 + 16 + col] = (_Float16)p1;
    }
    asm volatile("s_wait_dscnt 0" ::: "memory");   // LDS store -> reload as A frag
    const v16h ap = load_a_f16(pbuf, 32, lane);    // P 16x32
#pragma unroll
    for (int dt = 0; dt < 4; ++dt) {
      v16h bv = load_b_vt(Vt + (size_t)(dt * 16) * SEQ + key0, lane);
      o[dt] = wmma_f16(ap, bv, o[dt]);             // O += P @ V
    }
  }
  // epilogue: O /= l, store merged heads at [b*SEQ+q, h*64 + d] (f16)
  const int rbase = (lane >> 4) << 3;
  const int cn = lane & 15;
#pragma unroll
  for (int dt = 0; dt < 4; ++dt) {
#pragma unroll
    for (int r = 0; r < 8; ++r) {
      const int qq = q0 + rbase + r;
      const float val = o[dt][r] / lrow[r];
      ao[((size_t)bb * SEQ + qq) * DH + h * 64 + dt * 16 + cn] = (_Float16)val;
    }
  }
}

// ------ kernel 3: output projection + bias (all-f16 GEMM, 32x64/wave) ------
__global__ void __launch_bounds__(128) out_proj_kernel(
    const _Float16* __restrict__ ao, const _Float16* __restrict__ Wfc,
    const float* __restrict__ bias, float* __restrict__ out) {
  const int lane = threadIdx.x & 31;
  const int wave = threadIdx.x >> 5;
  const int task = blockIdx.x * 4 + wave;          // 256 * 12 tasks
  const int NG = DIM / 64;                         // 12
  const int mt = task / NG;
  const int n0 = (task % NG) * 64;
  const int row0 = mt * 32;

  v8f acc[2][4] = {};
  for (int k0 = 0; k0 < DH; k0 += 32) {
    const v16h a0 = load_a_f16(ao + (size_t)row0 * DH + k0, DH, lane);
    const v16h a1 = load_a_f16(ao + (size_t)(row0 + 16) * DH + k0, DH, lane);
#pragma unroll
    for (int nt = 0; nt < 4; ++nt) {
      const v16h b = load_b_f16(Wfc + (size_t)(n0 + nt * 16) * DH + k0, DH, lane);
      acc[0][nt] = wmma_f16(a0, b, acc[0][nt]);
      acc[1][nt] = wmma_f16(a1, b, acc[1][nt]);
    }
  }
  const int rbase = (lane >> 4) << 3;
  const int cn = lane & 15;
#pragma unroll
  for (int nt = 0; nt < 4; ++nt) {
    const int col = n0 + nt * 16 + cn;
    const float bv = bias[col];
#pragma unroll
    for (int rs = 0; rs < 2; ++rs)
#pragma unroll
      for (int r = 0; r < 8; ++r)
        out[(size_t)(row0 + rs * 16 + rbase + r) * DIM + col] = acc[rs][nt][r] + bv;
  }
}

extern "C" void kernel_launch(void* const* d_in, const int* in_sizes, int n_in,
                              void* d_out, int out_size, void* d_ws, size_t ws_size,
                              hipStream_t stream) {
  const float* img  = (const float*)d_in[0];   // [8,1024,768]
  const float* Wqkv = (const float*)d_in[1];   // [2304,768]
  const float* Wfc  = (const float*)d_in[2];   // [768,768]
  const float* bfc  = (const float*)d_in[3];   // [768]
  float* out = (float*)d_out;                  // [8,1024,768]

  const size_t per     = (size_t)BATCH * NUM_HEADS * SEQ * DIM_HEADS;  // 6291456
  const size_t n_img   = (size_t)BATCH * SEQ * DIM;                    // 6291456
  const size_t n_wqkv  = (size_t)QKV_OUT * DIM;                        // 1769472
  const size_t n_wfc   = (size_t)DIM * DH;                             //  589824

  _Float16* q16    = (_Float16*)d_ws;
  _Float16* k16    = q16 + per;
  _Float16* vt16   = k16 + per;          // transposed V: [B,H,64,SEQ]
  _Float16* ao     = vt16 + per;         // [8192, 768]
  _Float16* img16  = ao + per;
  _Float16* wqkv16 = img16 + n_img;
  _Float16* wfc16  = wqkv16 + n_wqkv;

  // one-shot fp32 -> fp16 (all counts divisible by 1024 floats/block)
  cvt_f32_f16<<<n_img  / 1024, 256, 0, stream>>>(img,  img16);
  cvt_f32_f16<<<n_wqkv / 1024, 256, 0, stream>>>(Wqkv, wqkv16);
  cvt_f32_f16<<<n_wfc  / 1024, 256, 0, stream>>>(Wfc,  wfc16);

  // 256 row-supertiles x 36 col-groups, 4 waves/block
  qkv_proj_kernel<<<(256 * 36) / 4, 128, 0, stream>>>(img16, wqkv16, q16, k16, vt16);
  // 8*12*64 wave-tasks, 4 waves/block
  attn_kernel<<<(BATCH * NUM_HEADS * (SEQ / 16)) / 4, 128, 0, stream>>>(q16, k16, vt16, ao);
  // 256 row-supertiles x 12 col-groups, 4 waves/block
  out_proj_kernel<<<(256 * 12) / 4, 128, 0, stream>>>(ao, wfc16, bfc, out);
}
